// PyTorchBEATsAttention_65936337928800
// MI455X (gfx1250) — compile-verified, matrix operand order checked by reading, not run on hardware
//
#include <hip/hip_runtime.h>
#include <hip/hip_bf16.h>

typedef __bf16 bf16;
typedef __attribute__((ext_vector_type(8)))  bf16  v8bf;
typedef __attribute__((ext_vector_type(16))) bf16  v16bf;
typedef __attribute__((ext_vector_type(8)))  float v8f;
typedef int v4i_vs __attribute__((vector_size(16)));   // matches builtin's V4i

#define B_  4
#define S_  2048
#define E_  768
#define H_  12
#define D_  64
#define KT_ 24    // 768/32 k-tiles
#define NT_ 48    // 768/16 n-tiles
#define MT_ 512   // 8192/16 m-tiles

#if __has_builtin(__builtin_amdgcn_global_load_async_to_lds_b128)
#define HAVE_ASYNC 1
#else
#define HAVE_ASYNC 0
#endif

__device__ __forceinline__ v8f wmma_bf16(v16bf a, v16bf b, v8f c) {
  return __builtin_amdgcn_wmma_f32_16x16x32_bf16(false, a, false, b, (short)0, c, false, false);
}

// 16B global -> LDS copy. Async (ASYNCcnt-tracked) when the builtin exists.
__device__ __forceinline__ void copy16_g2l(const bf16* g, bf16* l) {
#if HAVE_ASYNC
  __builtin_amdgcn_global_load_async_to_lds_b128(
      (__attribute__((address_space(1))) v4i_vs*)(g),
      (__attribute__((address_space(3))) v4i_vs*)(l), 0, 0);
#else
  *(int4*)l = *(const int4*)g;
#endif
}

__device__ __forceinline__ void wait_async_le2() {
#if HAVE_ASYNC
#if __has_builtin(__builtin_amdgcn_s_wait_asynccnt)
  __builtin_amdgcn_s_wait_asynccnt(2);
#else
  asm volatile("s_wait_asynccnt 0x2" ::: "memory");
#endif
#endif
}
__device__ __forceinline__ void wait_async_le0() {
#if HAVE_ASYNC
#if __has_builtin(__builtin_amdgcn_s_wait_asynccnt)
  __builtin_amdgcn_s_wait_asynccnt(0);
#else
  asm volatile("s_wait_asynccnt 0x0" ::: "memory");
#endif
#endif
}

// Load a 16-element A/B fragment whose two 8-element K-chunks are contiguous
// (chunk0 at p, chunk1 at p+16 elements) per ISA 7.12.2 16-bit layouts.
__device__ __forceinline__ v16bf load_frag(const bf16* p) {
  v8bf lo = *(const v8bf*)p;
  v8bf hi = *(const v8bf*)(p + 16);
  v16bf v;
#pragma unroll
  for (int i = 0; i < 8; ++i) { v[i] = lo[i]; v[8 + i] = hi[i]; }
  return v;
}

// ---- pack fp32 (M x 768) row-major into A-fragment layout (bf16) ----------
__global__ void pack_a_f32(const float* __restrict__ src, bf16* __restrict__ dst) {
  int tid  = blockIdx.x * blockDim.x + threadIdx.x;
  int lane = tid & 31;
  int frag = tid >> 5;                 // MT_*KT_ fragments
  int kt = frag % KT_, mt = frag / KT_;
  int half = lane >> 4, r = lane & 15;
  const float* p = src + (size_t)(mt * 16 + r) * E_ + kt * 32 + half * 8;
  v16bf v;
#pragma unroll
  for (int i = 0; i < 8; ++i) v[i]     = (bf16)p[i];
#pragma unroll
  for (int i = 0; i < 8; ++i) v[8 + i] = (bf16)p[16 + i];
  *(v16bf*)(dst + ((size_t)frag << 9) + lane * 16) = v;
}

// ---- pack fp32 weight (768 x 768, K x N row-major) into B-fragment layout --
__global__ void pack_b_f32(const float* __restrict__ W, bf16* __restrict__ dst) {
  int tid  = blockIdx.x * blockDim.x + threadIdx.x;
  int lane = tid & 31;
  int frag = tid >> 5;                 // NT_*KT_ fragments
  int kt = frag % KT_, nt = frag / KT_;
  int half = lane >> 4, n = lane & 15;
  int col = nt * 16 + n;
  int k0  = kt * 32 + half * 8;
  v16bf v;
#pragma unroll
  for (int i = 0; i < 8; ++i) v[i]     = (bf16)W[(size_t)(k0 + i) * E_ + col];
#pragma unroll
  for (int i = 0; i < 8; ++i) v[8 + i] = (bf16)W[(size_t)(k0 + 16 + i) * E_ + col];
  *(v16bf*)(dst + ((size_t)frag << 9) + lane * 16) = v;
}

// ---- WMMA GEMM: (8192 x 768) @ (768 x 768) + bias, mode-specific epilogue -
// mode 0: scaled bf16 out, (B,H,S,D) layout   (Q: scale = 1/256, K: scale=1)
// mode 2: bf16 out transposed (B,H,D,S)       (V)
// mode 3: fp32 out row-major (B*S, E)         (final projection)
__global__ void gemm_bf16_wmma(const bf16* __restrict__ A, const bf16* __restrict__ Bp,
                               const float* __restrict__ bias, float scale, int mode,
                               bf16* __restrict__ outB, float* __restrict__ outF) {
  int lane = threadIdx.x & 31, wv = threadIdx.x >> 5;
  int gt = blockIdx.x * 8 + wv;              // 24576 tiles
  int nt = gt % NT_, mt = gt / NT_;
  const bf16* ap = A  + (size_t)(mt * KT_) * 512 + lane * 16;
  const bf16* bp = Bp + (size_t)(nt * KT_) * 512 + lane * 16;
  v8f acc = {};
#pragma unroll 4
  for (int kt = 0; kt < KT_; ++kt) {
    v16bf a = *(const v16bf*)ap; ap += 512;
    v16bf b = *(const v16bf*)bp; bp += 512;
    acc = wmma_bf16(a, b, acc);
  }
  int half = lane >> 4, c = lane & 15;
  int col  = nt * 16 + c;
  float bc = bias[col];
#pragma unroll
  for (int e = 0; e < 8; ++e) {
    int   row = mt * 16 + e + 8 * half;
    float v   = acc[e] + bc;
    if (mode == 3) {
      outF[(size_t)row * E_ + col] = v;
    } else {
      int b_ = row >> 11, s = row & (S_ - 1);
      int h  = col >> 6,  d = col & (D_ - 1);
      if (mode == 2)
        outB[(((size_t)b_ * H_ + h) * D_ + d) * S_ + s] = (bf16)v;
      else
        outB[(((size_t)b_ * H_ + h) * S_ + s) * D_ + d] = (bf16)(v * scale);
    }
  }
}

// ---- gating MLP: gate[b,h,s] from q (stored scaled by 1/256) ---------------
__global__ void grep_gate(const bf16* __restrict__ q, const float* __restrict__ gw,
                          const float* __restrict__ gb, const float* __restrict__ ga,
                          float* __restrict__ gate) {
  int tid = blockIdx.x * blockDim.x + threadIdx.x;
  if (tid >= B_ * H_ * S_) return;
  int s = tid & (S_ - 1), bh = tid >> 11, h = bh % H_;
  const bf16* qr = q + ((size_t)bh * S_ + s) * D_;
  float acc[8] = {0, 0, 0, 0, 0, 0, 0, 0};
  for (int d = 0; d < D_; ++d) {
    float qv = (float)qr[d] * 256.0f;   // recover query_layer = x@Wq + bq
#pragma unroll
    for (int j = 0; j < 8; ++j) acc[j] += qv * gw[d * 8 + j];
  }
  float g0 = 0.f, g1 = 0.f;
#pragma unroll
  for (int j = 0; j < 4; ++j) { g0 += acc[j] + gb[j]; g1 += acc[4 + j] + gb[4 + j]; }
  g0 = 1.0f / (1.0f + __expf(-g0));
  g1 = 1.0f / (1.0f + __expf(-g1));
  gate[tid] = g0 * (g1 * ga[h] - 1.0f) + 2.0f;
}

// ---- relative-position bias table: tab[h][(j-i)+2047] ----------------------
__global__ void bias_table(const float* __restrict__ rel_emb, float* __restrict__ tab) {
  int t = blockIdx.x * blockDim.x + threadIdx.x;
  if (t >= H_ * 4095) return;
  int h = t / 4095, idx = t % 4095;
  int rel = idx - 2047;                 // rel = j - i
  const int nb = 160, me = 80;
  int bucket = rel > 0 ? nb : 0;
  int rp = rel < 0 ? -rel : rel;
  int add;
  if (rp < me) {
    add = rp;
  } else {
    int lg = me + (int)(__logf((float)rp / (float)me) / __logf(10.0f) * (float)(nb - me));
    add = lg < nb - 1 ? lg : nb - 1;
  }
  tab[h * 4095 + idx] = rel_emb[(bucket + add) * H_ + h];
}

// ---- flash attention -------------------------------------------------------
// One block per (b,h, 128-row Q band); 8 waves = 8 Q tiles of the same (b,h).
// K/V 32-key blocks are staged into LDS once per block via double-buffered
// async global->LDS copies, so K/V fetch traffic drops 8x vs per-wave loads.
__global__ void attn_fwd(const bf16* __restrict__ q, const bf16* __restrict__ k,
                         const bf16* __restrict__ vt, const float* __restrict__ gate,
                         const float* __restrict__ tab, float* __restrict__ out) {
  __shared__ __align__(16) bf16 kst[2][32][64];    // [buf][key][d]
  __shared__ __align__(16) bf16 vst[2][64][32];    // [buf][d][key]
  __shared__ __align__(16) bf16 pbuf[8][16][32];   // per-wave P staging
  int tid  = threadIdx.x;
  int lane = tid & 31, wv = tid >> 5;
  int blk = blockIdx.x;                  // 768 = B*H*(S/128)
  int qb = blk & 15, bh = blk >> 4;
  int h = bh % H_, b = bh / H_;
  int half = lane >> 4, r = lane & 15;
  int i0 = qb * 128 + wv * 16;

  const bf16* kb = k  + (size_t)bh * S_ * D_;
  const bf16* vb = vt + (size_t)bh * D_ * S_;

  // cooperative 4KB+4KB stage: each of 256 threads moves one 16B chunk of K
  // and one of V.
  auto stage = [&](int buf, int j0) {
    copy16_g2l(kb + (size_t)(j0 + (tid >> 3)) * D_ + (tid & 7) * 8,
               &kst[buf][tid >> 3][(tid & 7) * 8]);
    copy16_g2l(vb + (size_t)(tid >> 2) * S_ + j0 + (tid & 3) * 8,
               &vst[buf][tid >> 2][(tid & 3) * 8]);
  };

  // Q fragments for this wave's tile (registers, reused across key loop)
  v16bf qa[2];
#pragma unroll
  for (int kt = 0; kt < 2; ++kt)
    qa[kt] = load_frag(q + ((size_t)bh * S_ + i0 + r) * D_ + kt * 32 + half * 8);

  float grow[8];
#pragma unroll
  for (int e = 0; e < 8; ++e)
    grow[e] = gate[(size_t)bh * S_ + i0 + e + 8 * half];
  // tab index base: h*4095 + 2047 + (j - i), j = j0 + r, i = i0 + e + 8*half
  const float* tb = tab + h * 4095 + 2047 + r - i0 - 8 * half;

  float m[8], l[8];
  v8f acc[4] = {};
#pragma unroll
  for (int e = 0; e < 8; ++e) { m[e] = -1e30f; l[e] = 0.f; }

  stage(0, 0);
  for (int jb = 0; jb < 64; ++jb) {
    int j0 = jb * 32;
    if (jb < 63) { stage((jb + 1) & 1, j0 + 32); wait_async_le2(); }
    else         { wait_async_le0(); }
    __syncthreads();
    int cb = jb & 1;

    // scores for 32 keys from LDS-staged K: two 16-col C frags, 2 WMMAs each
    v16bf k00 = load_frag(&kst[cb][r][half * 8]);
    v16bf k01 = load_frag(&kst[cb][r][32 + half * 8]);
    v16bf k10 = load_frag(&kst[cb][16 + r][half * 8]);
    v16bf k11 = load_frag(&kst[cb][16 + r][32 + half * 8]);
    v8f s0 = {}; v8f s1 = {};
    s0 = wmma_bf16(qa[0], k00, s0);
    s0 = wmma_bf16(qa[1], k01, s0);
    s1 = wmma_bf16(qa[0], k10, s1);
    s1 = wmma_bf16(qa[1], k11, s1);

    float p0[8], p1[8], sc[8];
#pragma unroll
    for (int e = 0; e < 8; ++e) {
      float x0 = s0[e] + grow[e] * tb[j0 - e];
      float x1 = s1[e] + grow[e] * tb[j0 + 16 - e];
      float mx = fmaxf(x0, x1);
      mx = fmaxf(mx, __shfl_xor(mx, 1, 32));
      mx = fmaxf(mx, __shfl_xor(mx, 2, 32));
      mx = fmaxf(mx, __shfl_xor(mx, 4, 32));
      mx = fmaxf(mx, __shfl_xor(mx, 8, 32));
      float mn = fmaxf(m[e], mx);
      float s  = __expf(m[e] - mn);
      float e0 = __expf(x0 - mn), e1 = __expf(x1 - mn);
      float rs = e0 + e1;
      rs += __shfl_xor(rs, 1, 32);
      rs += __shfl_xor(rs, 2, 32);
      rs += __shfl_xor(rs, 4, 32);
      rs += __shfl_xor(rs, 8, 32);
      l[e] = l[e] * s + rs;
      m[e] = mn;
      sc[e] = s; p0[e] = e0; p1[e] = e1;
    }
#pragma unroll
    for (int t = 0; t < 4; ++t)
#pragma unroll
      for (int e = 0; e < 8; ++e) acc[t][e] *= sc[e];

    // relayout P: C-fragment -> A-fragment via per-wave LDS tile
#pragma unroll
    for (int e = 0; e < 8; ++e) {
      pbuf[wv][e + 8 * half][r]      = (bf16)p0[e];
      pbuf[wv][e + 8 * half][16 + r] = (bf16)p1[e];
    }
    v8bf plo = *(const v8bf*)&pbuf[wv][r][half * 8];
    v8bf phi = *(const v8bf*)&pbuf[wv][r][16 + half * 8];
    v16bf pa;
#pragma unroll
    for (int i = 0; i < 8; ++i) { pa[i] = plo[i]; pa[8 + i] = phi[i]; }

    // O += P(16x32) x V(32x16) for 4 d-tiles, V fragments from LDS
#pragma unroll
    for (int t = 0; t < 4; ++t) {
      v16bf vf = load_frag(&vst[cb][t * 16 + r][half * 8]);
      acc[t] = wmma_bf16(pa, vf, acc[t]);
    }
    __syncthreads();   // all waves done reading buf before it is re-staged
  }

  float* ob = out + (size_t)b * S_ * E_;
#pragma unroll
  for (int t = 0; t < 4; ++t)
#pragma unroll
    for (int e = 0; e < 8; ++e) {
      int row = i0 + e + 8 * half;
      int col = h * 64 + t * 16 + r;
      ob[(size_t)row * E_ + col] = acc[t][e] / l[e];
    }
}

extern "C" void kernel_launch(void* const* d_in, const int* in_sizes, int n_in,
                              void* d_out, int out_size, void* d_ws, size_t ws_size,
                              hipStream_t stream) {
  (void)in_sizes; (void)n_in; (void)out_size; (void)ws_size;
  const float* x       = (const float*)d_in[0];
  const float* Wq      = (const float*)d_in[1];
  const float* bq      = (const float*)d_in[2];
  const float* Wk      = (const float*)d_in[3];
  const float* bk      = (const float*)d_in[4];
  const float* Wv      = (const float*)d_in[5];
  const float* bv      = (const float*)d_in[6];
  const float* Wo      = (const float*)d_in[7];
  const float* bo      = (const float*)d_in[8];
  const float* rel_emb = (const float*)d_in[9];
  const float* grep_W  = (const float*)d_in[10];
  const float* grep_b  = (const float*)d_in[11];
  const float* grep_a  = (const float*)d_in[12];
  float* out = (float*)d_out;

  char* ws = (char*)d_ws;
  size_t off = 0;
  auto alloc = [&](size_t bytes) {
    void* p = ws + off;
    off += (bytes + 255) & ~(size_t)255;
    return p;
  };
  bf16* ApackX  = (bf16*)alloc((size_t)8192 * 768 * 2);
  bf16* ApackO  = (bf16*)alloc((size_t)8192 * 768 * 2);
  bf16* BpQ     = (bf16*)alloc((size_t)768 * 768 * 2);
  bf16* BpK     = (bf16*)alloc((size_t)768 * 768 * 2);
  bf16* BpV     = (bf16*)alloc((size_t)768 * 768 * 2);
  bf16* BpO     = (bf16*)alloc((size_t)768 * 768 * 2);
  bf16* qbuf    = (bf16*)alloc((size_t)B_ * H_ * S_ * D_ * 2);
  bf16* kbuf    = (bf16*)alloc((size_t)B_ * H_ * S_ * D_ * 2);
  bf16* vtbuf   = (bf16*)alloc((size_t)B_ * H_ * S_ * D_ * 2);
  float* gateb  = (float*)alloc((size_t)B_ * H_ * S_ * 4);
  float* tabb   = (float*)alloc((size_t)H_ * 4095 * 4);
  float* attnO  = (float*)alloc((size_t)8192 * 768 * 4);

  const float qscale = 0.125f / 32.0f;   // D^-0.5 / ALPHA = 1/256

  pack_a_f32<<<1536, 256, 0, stream>>>(x, ApackX);
  pack_b_f32<<<144, 256, 0, stream>>>(Wq, BpQ);
  pack_b_f32<<<144, 256, 0, stream>>>(Wk, BpK);
  pack_b_f32<<<144, 256, 0, stream>>>(Wv, BpV);
  pack_b_f32<<<144, 256, 0, stream>>>(Wo, BpO);

  gemm_bf16_wmma<<<3072, 256, 0, stream>>>(ApackX, BpQ, bq, qscale, 0, qbuf, nullptr);
  gemm_bf16_wmma<<<3072, 256, 0, stream>>>(ApackX, BpK, bk, 1.0f,   0, kbuf, nullptr);
  gemm_bf16_wmma<<<3072, 256, 0, stream>>>(ApackX, BpV, bv, 1.0f,   2, vtbuf, nullptr);

  grep_gate<<<384, 256, 0, stream>>>(qbuf, grep_W, grep_b, grep_a, gateb);
  bias_table<<<192, 256, 0, stream>>>(rel_emb, tabb);

  attn_fwd<<<768, 256, 0, stream>>>(qbuf, kbuf, vtbuf, gateb, tabb, attnO);

  pack_a_f32<<<1536, 256, 0, stream>>>(attnO, ApackO);
  gemm_bf16_wmma<<<3072, 256, 0, stream>>>(ApackO, BpO, bo, 1.0f, 3, nullptr, out);
}
